// SymplecticIntegrator_18648747999579
// MI455X (gfx1250) — compile-verified, matrix operand order checked by reading, not run on hardware
//
#include <hip/hip_runtime.h>
#include <cstdint>
#include <cstddef>

typedef __attribute__((ext_vector_type(16))) __bf16 v16bf;
typedef __attribute__((ext_vector_type(8)))  float  v8f;

namespace {

constexpr int   kD         = 256;
constexpr int   kR         = 64;
constexpr int   kWaves     = 8;
constexpr int   kRowsPerWG = kWaves * 16;
constexpr float kDt        = 0.01f;

constexpr float kPi       = 3.14159265358979323846f;
constexpr float kTwoPi    = 6.28318530717958647692f;
constexpr float kInvTwoPi = 0.15915494309189533577f;

// LDS layout (bytes)
constexpr int kUOff     = 0;                              // 32KB swizzled bf16 U fragments
constexpr int kWOff     = 32 * 1024;                      // 32KB swizzled bf16 W fragments
constexpr int kVstOff   = 64 * 1024;                      // 8 waves * 8KB bf16 v staging
constexpr int kXstOff   = kVstOff + kWaves * 8 * 1024;    // 8 waves * 8KB bf16 x staging
constexpr int kSstOff   = kXstOff + kWaves * 8 * 1024;    // 8 waves * 2KB bf16 s staging
constexpr int kLdsBytes = kSstOff + kWaves * 2 * 1024;    // 212992 B = 208KB

// Wave-private LDS staging only needs compiler-level ordering: LDS ops from one
// wave complete in order on CDNA5, so this fence emits (nearly) nothing but
// pins the cross-lane ds_store -> ds_load ordering in the IR.
__device__ __forceinline__ void wave_fence() {
  __builtin_amdgcn_fence(__ATOMIC_ACQ_REL, "wavefront");
}

// Position of element (kin in 0..31, nin in 0..15) inside a 512-bf16 B-fragment
// chunk, matching the 16-bit WMMA operand VGPR layout
// (lanes 0-15: K 0-7,16-23; lanes 16-31: K 8-15,24-31).
__device__ __forceinline__ int frag_pos(int kin, int nin) {
  const int lane = nin + (((kin >> 3) & 1) << 4);
  const int r    = ((kin & 7) >> 1) + ((kin >> 4) << 2);
  const int half = kin & 1;
  return (lane * 16) + (r * 2) + half;
}

// A-fragment (16x32 bf16) from row-major bf16 staging: two ds_load_b128 per lane.
__device__ __forceinline__ v16bf load_a_frag(const __bf16* stage, int strideBf,
                                             int kbase, int lane) {
  const int m  = lane & 15;
  const int hi = (lane >> 4) & 1;
  const __bf16* p = stage + m * strideBf + kbase + hi * 8;
  union { uint4 u[2]; v16bf v; } un;
  un.u[0] = *(const uint4*)(p);
  un.u[1] = *(const uint4*)(p + 16);
  return un.v;
}

// B-fragment (32x16 bf16) from a pre-swizzled 1KB chunk: 8 contiguous dwords/lane.
__device__ __forceinline__ v16bf load_b_frag(const uint32_t* chunk, int lane) {
  union { uint4 u[2]; v16bf v; } un;
  const uint4* p = (const uint4*)(chunk + lane * 8);
  un.u[0] = p[0];
  un.u[1] = p[1];
  return un.v;
}

__device__ __forceinline__ v8f wmma_bf16(v16bf a, v16bf b, v8f c) {
  // (neg_a, A, neg_b, B, c_mod, C, reuse_a, reuse_b)
  return __builtin_amdgcn_wmma_f32_16x16x32_bf16(false, a, false, b, (short)0, c,
                                                 false, false);
}

__device__ __forceinline__ float dev_tanh(float x) {
#if __has_builtin(__builtin_amdgcn_tanhf)
  return __builtin_amdgcn_tanhf(x);          // v_tanh_f32 (gfx1250)
#else
  const float ax = __builtin_fabsf(x);
  const float e  = __expf(2.0f * ax);        // v_exp_f32 path; e==inf -> t==1
  const float t  = 1.0f - 2.0f / (e + 1.0f);
  return __builtin_copysignf(t, x);
#endif
}

__device__ __forceinline__ float wrap_pi(float x) {
  float t = x + kPi;
  t = t - floorf(t * kInvTwoPi) * kTwoPi;
  return t - kPi;
}

// out[4] = (stage[16xD] @ U) tiles, accumulated in f32.
__device__ __forceinline__ void gemm_rU(const __bf16* ast, const uint32_t* Uch,
                                        int lane, v8f out[4]) {
  out[0] = {}; out[1] = {}; out[2] = {}; out[3] = {};
  #pragma unroll
  for (int kt = 0; kt < 8; ++kt) {
    const v16bf a = load_a_frag(ast, kD, kt * 32, lane);
    #pragma unroll
    for (int rt = 0; rt < 4; ++rt)
      out[rt] = wmma_bf16(a, load_b_frag(Uch + (kt * 4 + rt) * 256, lane), out[rt]);
  }
}

// h = vst@U, g = xst@U with the shared U fragment feeding both WMMAs.
__device__ __forceinline__ void gemm_rU2(const __bf16* vst, const __bf16* xst,
                                         const uint32_t* Uch, int lane,
                                         v8f h[4], v8f g[4]) {
  h[0] = {}; h[1] = {}; h[2] = {}; h[3] = {};
  g[0] = {}; g[1] = {}; g[2] = {}; g[3] = {};
  #pragma unroll
  for (int kt = 0; kt < 8; ++kt) {
    const v16bf av = load_a_frag(vst, kD, kt * 32, lane);
    const v16bf ax = load_a_frag(xst, kD, kt * 32, lane);
    #pragma unroll
    for (int rt = 0; rt < 4; ++rt) {
      const v16bf bu = load_b_frag(Uch + (kt * 4 + rt) * 256, lane);
      h[rt] = wmma_bf16(av, bu, h[rt]);
      g[rt] = wmma_bf16(ax, bu, g[rt]);
    }
  }
}

// vr += scale * (force - (h*h*gt) @ W)
__device__ __forceinline__ void kick(const v8f h[4], const v8f gt[4], __bf16* sst,
                                     const uint32_t* Wch,
                                     const float* __restrict__ force,
                                     long rowBase, int m0, int ncol, int lane,
                                     v8f* vr, float scale) {
  #pragma unroll
  for (int rt = 0; rt < 4; ++rt) {
    #pragma unroll
    for (int r = 0; r < 8; ++r) {
      const float hh = h[rt][r];
      sst[(m0 + r) * kR + rt * 16 + ncol] = (__bf16)(hh * hh * gt[rt][r]);
    }
  }
  wave_fence();
  const v16bf as0 = load_a_frag(sst, kR, 0, lane);
  const v16bf as1 = load_a_frag(sst, kR, 32, lane);
  #pragma unroll
  for (int nt = 0; nt < 16; ++nt) {
    v8f acc = {};
    acc = wmma_bf16(as0, load_b_frag(Wch + (nt * 2 + 0) * 256, lane), acc);
    acc = wmma_bf16(as1, load_b_frag(Wch + (nt * 2 + 1) * 256, lane), acc);
    #pragma unroll
    for (int r = 0; r < 8; ++r) {
      const float f = force[(rowBase + m0 + r) * (long)kD + nt * 16 + ncol];
      vr[nt][r] += scale * (f - acc[r]);
    }
  }
}

}  // namespace

extern "C" __global__ void __launch_bounds__(256, 1)
sympl_wmma_kernel(const float* __restrict__ x_in, const float* __restrict__ v_in,
                  const float* __restrict__ force, const float* __restrict__ U,
                  const float* __restrict__ W, const int* __restrict__ steps_p,
                  float* __restrict__ x_out, float* __restrict__ v_out)
{
  extern __shared__ char smem[];
  const int tid  = threadIdx.x;
  const int lane = tid & 31;
  const int wave = tid >> 5;

  __bf16* vst = (__bf16*)(smem + kVstOff) + wave * (16 * kD);
  __bf16* xst = (__bf16*)(smem + kXstOff) + wave * (16 * kD);
  __bf16* sst = (__bf16*)(smem + kSstOff) + wave * (16 * kR);
  const uint32_t* Uch = (const uint32_t*)(smem + kUOff);
  const uint32_t* Wch = (const uint32_t*)(smem + kWOff);

  // Convert U, W to bf16 and pre-swizzle into exact B-fragment layout.
  {
    __bf16* ub = (__bf16*)(smem + kUOff);
    for (int idx = tid; idx < kD * kR; idx += 256) {     // U is [D][R], row = K
      const int K = idx >> 6, N = idx & 63;
      const int chunk = (K >> 5) * 4 + (N >> 4);         // [kstep][rtile]
      ub[chunk * 512 + frag_pos(K & 31, N & 15)] = (__bf16)U[idx];
    }
    __bf16* wb = (__bf16*)(smem + kWOff);
    for (int idx = tid; idx < kR * kD; idx += 256) {     // W is [R][D], row = K
      const int K = idx >> 8, N = idx & 255;
      const int chunk = (N >> 4) * 2 + (K >> 5);         // [ntile][kstep]
      wb[chunk * 512 + frag_pos(K & 31, N & 15)] = (__bf16)W[idx];
    }
  }

  const long rowBase = (long)blockIdx.x * kRowsPerWG + wave * 16;
  const int  m0      = ((lane >> 4) & 1) * 8;   // C-layout: M = m0 + vgpr_index
  const int  ncol    = lane & 15;               // C-layout: N = 16*tile + ncol

  // x, v resident in registers in WMMA C-layout: 16 column-tiles of v8f each.
  v8f xr[16], vr[16];
  #pragma unroll
  for (int nt = 0; nt < 16; ++nt) {
    #pragma unroll
    for (int r = 0; r < 8; ++r) {
      const long off = (rowBase + m0 + r) * (long)kD + nt * 16 + ncol;
      xr[nt][r] = x_in[off];
      vr[nt][r] = v_in[off];
    }
  }
  // Initial bf16 staging of x, v.
  #pragma unroll
  for (int nt = 0; nt < 16; ++nt) {
    #pragma unroll
    for (int r = 0; r < 8; ++r) {
      const int sidx = (m0 + r) * kD + nt * 16 + ncol;
      vst[sidx] = (__bf16)vr[nt][r];
      xst[sidx] = (__bf16)xr[nt][r];
    }
  }

  // Single block-wide barrier: U/W fragments (cross-wave) become visible.
  // All later staging is wave-private; wave_fence() suffices from here on.
  __syncthreads();

  const int   nsteps = *steps_p;
  const float halfdt = 0.5f * kDt;

  // gt = tanh(x @ U): computed once, then refreshed only when x changes.
  v8f gt[4];
  {
    v8f g[4];
    gemm_rU(xst, Uch, lane, g);
    #pragma unroll
    for (int rt = 0; rt < 4; ++rt)
      #pragma unroll
      for (int r = 0; r < 8; ++r)
        gt[rt][r] = dev_tanh(g[rt][r]);
  }

  for (int s = 0; s < nsteps; ++s) {
    // ---- first half-kick: reuses gt from the previous x ----
    v8f h[4];
    gemm_rU(vst, Uch, lane, h);
    kick(h, gt, sst, Wch, force, rowBase, m0, ncol, lane, vr, halfdt);  // vr=v_half

    // ---- drift: x = wrap(x + dt*v_half); restage v_half and new x ----
    #pragma unroll
    for (int nt = 0; nt < 16; ++nt) {
      #pragma unroll
      for (int r = 0; r < 8; ++r) {
        const int sidx = (m0 + r) * kD + nt * 16 + ncol;
        vst[sidx] = (__bf16)vr[nt][r];
        const float xn = wrap_pi(xr[nt][r] + kDt * vr[nt][r]);
        xr[nt][r] = xn;
        xst[sidx] = (__bf16)xn;
      }
    }
    wave_fence();

    // ---- second half-kick: h and fresh g share U fragments ----
    v8f g[4];
    gemm_rU2(vst, xst, Uch, lane, h, g);
    #pragma unroll
    for (int rt = 0; rt < 4; ++rt)
      #pragma unroll
      for (int r = 0; r < 8; ++r)
        gt[rt][r] = dev_tanh(g[rt][r]);
    kick(h, gt, sst, Wch, force, rowBase, m0, ncol, lane, vr, halfdt);  // vr=v_next

    #pragma unroll
    for (int nt = 0; nt < 16; ++nt) {
      #pragma unroll
      for (int r = 0; r < 8; ++r)
        vst[(m0 + r) * kD + nt * 16 + ncol] = (__bf16)vr[nt][r];
    }
    wave_fence();
  }

  #pragma unroll
  for (int nt = 0; nt < 16; ++nt) {
    #pragma unroll
    for (int r = 0; r < 8; ++r) {
      const long off = (rowBase + m0 + r) * (long)kD + nt * 16 + ncol;
      x_out[off] = xr[nt][r];
      v_out[off] = vr[nt][r];
    }
  }
}

extern "C" void kernel_launch(void* const* d_in, const int* in_sizes, int n_in,
                              void* d_out, int out_size, void* d_ws, size_t ws_size,
                              hipStream_t stream) {
  const float* x_in  = (const float*)d_in[0];
  const float* v_in  = (const float*)d_in[1];
  const float* force = (const float*)d_in[2];
  const float* U     = (const float*)d_in[3];
  const float* W     = (const float*)d_in[4];
  const int*   steps = (const int*)d_in[5];

  const int rows = in_sizes[0] / kD;          // 65536
  float* x_out = (float*)d_out;
  float* v_out = x_out + (size_t)rows * kD;

  hipFuncSetAttribute(reinterpret_cast<const void*>(sympl_wmma_kernel),
                      hipFuncAttributeMaxDynamicSharedMemorySize, kLdsBytes);

  dim3 grid(rows / kRowsPerWG);               // 512 workgroups
  sympl_wmma_kernel<<<grid, dim3(256), kLdsBytes, stream>>>(
      x_in, v_in, force, U, W, steps, x_out, v_out);
}
